// S3FLoss_39702677684806
// MI455X (gfx1250) — compile-verified, matrix operand order checked by reading, not run on hardware
//
#include <hip/hip_runtime.h>

// ---------------------------------------------------------------------------
// S3F loss for gfx1250 (MI455X, wave32).
//  - prep:     focal terms + dice sums (WMMA f16 hi/lo tensor reduction) +
//              DT cost-volume init + mask-nonempty flags
//  - dt_pass:  exact squared EDT, separable min-plus over each axis, in-place
//              (each thread owns one 64-elem line in registers)
//  - boundary: (sigmoid(x)-t)^2 * (field_p^2 + field_t^2), WMMA reduction
//  - finalize: combine partials -> scalar loss
// Workspace: 16 * 64^3 f32 DT volumes (16 MB) + 8 flag ints + 5*512 partials.
// ---------------------------------------------------------------------------

typedef __attribute__((ext_vector_type(16))) _Float16 v16h;
typedef __attribute__((ext_vector_type(8)))  float    v8f;

#define S_VOL   262144      // 64*64*64
#define N_TOT   1048576     // 2*2*64^3
#define NBLK    512         // N_TOT / (256 threads * 8 elems)
#define BIG_F   1e12f
#define EPS_F   1e-7f

__device__ __forceinline__ float sigmoidf_(float x) {
    return 1.0f / (1.0f + expf(-x));
}
// numerically stable log(sigmoid(z))
__device__ __forceinline__ float logsigf_(float z) {
    return fminf(z, 0.0f) - log1pf(expf(-fabsf(z)));
}

__device__ __forceinline__ void load8(const float* __restrict__ p, float* v) {
    float4 a = *(const float4*)(p);
    float4 b = *(const float4*)(p + 4);
    v[0]=a.x; v[1]=a.y; v[2]=a.z; v[3]=a.w;
    v[4]=b.x; v[5]=b.y; v[6]=b.z; v[7]=b.w;
}
__device__ __forceinline__ void store8(float* __restrict__ p, const float* v) {
    *(float4*)(p)     = make_float4(v[0], v[1], v[2], v[3]);
    *(float4*)(p + 4) = make_float4(v[4], v[5], v[6], v[7]);
}

// Tensor-core wave reduction: 32 lanes x 8 f32 values -> accumulate into C.
// Each value is split into an exact hi/lo f16 pair; B is all-ones so the
// f16*1.0 products are exact and accumulation happens in f32.
__device__ __forceinline__ void wmma_sum8(v8f& c, const float* v) {
    v16h a;
#pragma unroll
    for (int j = 0; j < 8; ++j) {
        float x = v[j];
        _Float16 hi = (_Float16)x;
        _Float16 lo = (_Float16)(x - (float)hi);
        a[2*j]   = hi;
        a[2*j+1] = lo;
    }
    v16h b;
#pragma unroll
    for (int j = 0; j < 16; ++j) b[j] = (_Float16)1.0f;
    c = __builtin_amdgcn_wmma_f32_16x16x32_f16(
            /*neg_a=*/false, a, /*neg_b=*/false, b,
            /*c_mod=*/(short)0, c, /*reuse_a=*/false, /*reuse_b=*/false);
}
// D[m][n]: lanes 0..15 hold rows 0..7 in vgprs 0..7, lanes 16..31 rows 8..15.
__device__ __forceinline__ float wmma_total(v8f c) {
    float s = 0.0f;
#pragma unroll
    for (int j = 0; j < 8; ++j) s += c[j];
    s += __shfl_xor(s, 16, 32);
    return s;
}

__global__ void s3f_zero_flags(int* __restrict__ flags) {
    if (threadIdx.x < 8) flags[threadIdx.x] = 0;
}

__global__ void __launch_bounds__(256) s3f_prep(
    const float* __restrict__ pred, const float* __restrict__ targ,
    float* __restrict__ dt, int* __restrict__ flags, float* __restrict__ part)
{
    const int tid  = threadIdx.x;
    const int blk  = blockIdx.x;
    const int base = blk * 2048 + tid * 8;
    const int slab = blk >> 7;            // 128 blocks per (b,c) slab
    const int idx  = base - slab * S_VOL; // offset within slab

    float x[8], t[8];
    load8(pred + base, x);
    load8(targ + base, t);

    float vf[8], vi[8], vp[8], vt[8];
    float pf[8], pb[8], tf[8], tb[8];
    bool anypm = false, anytm = false;
#pragma unroll
    for (int j = 0; j < 8; ++j) {
        const float xx = x[j], tt = t[j];
        const float s  = sigmoidf_(xx);
        // focal (sigmoid focal, gamma=2) with uncertainty weight
        const float ce   = xx - xx*tt - logsigf_(xx);
        const float zmod = -xx * (2.0f*tt - 1.0f);
        const float mod  = expf(2.0f * logsigf_(zmod));
        const float p    = fminf(fmaxf(s, EPS_F), 1.0f - EPS_F);
        const float wu   = 1.0f - fabsf(p - 0.5f);
        vf[j] = mod * ce * wu;
        // dice/jaccard sums
        vi[j] = s * tt;
        vp[j] = s * s;
        vt[j] = tt * tt;
        // EDT cost volumes: edt(m) seeds are the ZEROS of the cost field
        const bool pm = xx > 0.0f;   // sigmoid(x) > 0.5
        const bool tm = tt > 0.5f;
        pf[j] = pm ? BIG_F : 0.0f;   // where(mask, BIG, 0)
        pb[j] = pm ? 0.0f : BIG_F;   // where(mask, 0, BIG)
        tf[j] = tm ? BIG_F : 0.0f;
        tb[j] = tm ? 0.0f : BIG_F;
        anypm |= pm; anytm |= tm;
    }
    store8(dt + (0*4 + slab)*S_VOL + idx, pf);
    store8(dt + (1*4 + slab)*S_VOL + idx, pb);
    store8(dt + (2*4 + slab)*S_VOL + idx, tf);
    store8(dt + (3*4 + slab)*S_VOL + idx, tb);
    if (anypm) atomicOr(flags + slab, 1);
    if (anytm) atomicOr(flags + 4 + slab, 1);

    v8f cf = {0,0,0,0,0,0,0,0};
    v8f ci = cf, cp = cf, ct = cf;
    wmma_sum8(cf, vf);
    wmma_sum8(ci, vi);
    wmma_sum8(cp, vp);
    wmma_sum8(ct, vt);
    const float sf = wmma_total(cf);
    const float si = wmma_total(ci);
    const float sp = wmma_total(cp);
    const float st = wmma_total(ct);

    __shared__ float wsum[8][4];
    const int wid = tid >> 5, lane = tid & 31;
    if (lane == 0) {
        wsum[wid][0] = sf; wsum[wid][1] = si;
        wsum[wid][2] = sp; wsum[wid][3] = st;
    }
    __syncthreads();
    if (tid == 0) {
        float q0=0.f, q1=0.f, q2=0.f, q3=0.f;
#pragma unroll
        for (int w = 0; w < 8; ++w) {
            q0 += wsum[w][0]; q1 += wsum[w][1];
            q2 += wsum[w][2]; q3 += wsum[w][3];
        }
        part[blk]          = q0;
        part[NBLK   + blk] = q1;
        part[2*NBLK + blk] = q2;
        part[3*NBLK + blk] = q3;
    }
}

// One separable min-plus EDT pass: out[x] = min_y f[y] + (x-y)^2
// rewritten as x^2 + min_y (h[y] - 2x*y) with h[y] = f[y] + y^2
// (1 fma + 1 min per unrolled inner step). One thread per 64-elem line.
__global__ void __launch_bounds__(256) s3f_dt_pass(float* __restrict__ dt, int axis)
{
    const int gid = blockIdx.x * 256 + threadIdx.x;  // 16 vols * 4096 lines
    const int vol = gid >> 12;
    const int rem = gid & 4095;
    const int a = rem >> 6, b = rem & 63;
    int base, stride;
    if (axis == 0)      { base = a*64   + b;    stride = 4096; }
    else if (axis == 1) { base = a*4096 + b;    stride = 64;   }
    else                { base = a*4096 + b*64; stride = 1;    }
    float* line = dt + vol * S_VOL + base;

    float h[64];
#pragma unroll
    for (int k = 0; k < 64; ++k) h[k] = line[k*stride] + (float)(k*k);
    for (int xp = 0; xp < 64; ++xp) {
        const float t2x = 2.0f * (float)xp;
        float m = h[0];
#pragma unroll
        for (int y = 1; y < 64; ++y)
            m = fminf(m, fmaf(-t2x, (float)y, h[y]));
        line[xp*stride] = m + (float)(xp*xp);
    }
}

__global__ void __launch_bounds__(256) s3f_boundary(
    const float* __restrict__ pred, const float* __restrict__ targ,
    const float* __restrict__ dt, const int* __restrict__ flags,
    float* __restrict__ bpart)
{
    const int tid  = threadIdx.x;
    const int blk  = blockIdx.x;
    const int base = blk * 2048 + tid * 8;
    const int slab = blk >> 7;
    const int idx  = base - slab * S_VOL;

    float x[8], t[8], pf[8], pb[8], tf[8], tb[8];
    load8(pred + base, x);
    load8(targ + base, t);
    load8(dt + (0*4 + slab)*S_VOL + idx, pf);
    load8(dt + (1*4 + slab)*S_VOL + idx, pb);
    load8(dt + (2*4 + slab)*S_VOL + idx, tf);
    load8(dt + (3*4 + slab)*S_VOL + idx, tb);
    const float fp_on = flags[slab]     ? 1.0f : 0.0f;
    const float ft_on = flags[4 + slab] ? 1.0f : 0.0f;

    float term[8];
#pragma unroll
    for (int j = 0; j < 8; ++j) {
        const float s = sigmoidf_(x[j]);
        float e = s - t[j];
        e = e * e;
        const float Fp = (sqrtf(pf[j]) + sqrtf(pb[j])) * fp_on;
        const float Ft = (sqrtf(tf[j]) + sqrtf(tb[j])) * ft_on;
        term[j] = e * (Fp*Fp + Ft*Ft);   // HD_ALPHA = 2
    }

    v8f c = {0,0,0,0,0,0,0,0};
    wmma_sum8(c, term);
    const float s = wmma_total(c);

    __shared__ float wsum[8];
    const int wid = tid >> 5, lane = tid & 31;
    if (lane == 0) wsum[wid] = s;
    __syncthreads();
    if (tid == 0) {
        float q = 0.f;
#pragma unroll
        for (int w = 0; w < 8; ++w) q += wsum[w];
        bpart[blk] = q;
    }
}

__global__ void __launch_bounds__(256) s3f_finalize(
    const float* __restrict__ part, float* __restrict__ out)
{
    __shared__ float red[256];
    __shared__ float acc[2];
    __shared__ float dice[4];
    const int tid = threadIdx.x;

    // focal total
    red[tid] = part[tid] + part[tid + 256];
    __syncthreads();
    for (int s = 128; s > 0; s >>= 1) {
        if (tid < s) red[tid] += red[tid + s];
        __syncthreads();
    }
    if (tid == 0) acc[0] = red[0];
    __syncthreads();

    // boundary total
    red[tid] = part[4*NBLK + tid] + part[4*NBLK + tid + 256];
    __syncthreads();
    for (int s = 128; s > 0; s >>= 1) {
        if (tid < s) red[tid] += red[tid + s];
        __syncthreads();
    }
    if (tid == 0) acc[1] = red[0];
    __syncthreads();

    // per-channel dice (jaccard, squared_pred): 128 partials per slab
    if (tid < 4) {
        float it = 0.f, p2 = 0.f, t2 = 0.f;
        for (int k = 0; k < 128; ++k) {
            const int i = tid * 128 + k;
            it += part[NBLK   + i];
            p2 += part[2*NBLK + i];
            t2 += part[3*NBLK + i];
        }
        const float denom = 2.0f * ((p2 + t2) - it);
        dice[tid] = 1.0f - (2.0f * it) / (denom + 1e-5f);
    }
    __syncthreads();
    if (tid == 0) {
        const float structure = 0.25f * (dice[0] + dice[1] + dice[2] + dice[3]);
        const float invN = 1.0f / (float)N_TOT;
        // ALPHA*focal + BETA*structure + DELTA*boundary
        out[0] = 1.0f * (acc[0] * invN) + 0.5f * structure + 0.3f * (acc[1] * invN);
    }
}

extern "C" void kernel_launch(void* const* d_in, const int* in_sizes, int n_in,
                              void* d_out, int out_size, void* d_ws, size_t ws_size,
                              hipStream_t stream) {
    const float* pred = (const float*)d_in[0];
    const float* targ = (const float*)d_in[1];
    float* out = (float*)d_out;
    char*  ws  = (char*)d_ws;

    float* dtv   = (float*)ws;                                  // 16 * S_VOL f32
    int*   flags = (int*)(ws + (size_t)16 * S_VOL * 4);         // 8 ints
    float* part  = (float*)(ws + (size_t)16 * S_VOL * 4 + 64);  // 5*NBLK f32

    s3f_zero_flags<<<1, 32, 0, stream>>>(flags);
    s3f_prep<<<NBLK, 256, 0, stream>>>(pred, targ, dtv, flags, part);
    s3f_dt_pass<<<256, 256, 0, stream>>>(dtv, 0);
    s3f_dt_pass<<<256, 256, 0, stream>>>(dtv, 1);
    s3f_dt_pass<<<256, 256, 0, stream>>>(dtv, 2);
    s3f_boundary<<<NBLK, 256, 0, stream>>>(pred, targ, dtv, flags, part + 4*NBLK);
    s3f_finalize<<<1, 256, 0, stream>>>(part, out);
}
